// LIIF_42872363549082
// MI455X (gfx1250) — matrix operand order repaired and checked by reading, not compiled, not run on hardware
//
#include <hip/hip_runtime.h>

// ---------------------------------------------------------------------------
// LIIF forward for MI455X (gfx1250): f16 WMMA MLP, wave32.
// 2x2 register blocking: each wave owns TWO 16-row M-tiles (32 rows = 8
// queries x 4 shifts) and processes TWO 16-col N-tiles at a time, so every
// pair of A/B fragments feeds FOUR v_wmma ops (64 FLOP per L2 byte on the
// weight stream, halved LDS traffic on the activation stream).
// ---------------------------------------------------------------------------

typedef __attribute__((ext_vector_type(16))) _Float16 v16h;
typedef __attribute__((ext_vector_type(8)))  _Float16 v8h;
typedef __attribute__((ext_vector_type(8)))  float    v8f;

#define HID    256
#define XROW   608            // 596 padded to 19*32
#define KT0    19             // k-tiles for layer 0
#define KTH    8              // k-tiles for hidden layers (256/32)
#define NT_H   16             // n-tiles for hidden layers (256/16)
#define WAVES  4
#define MT     2              // M-tiles per wave (32 rows)
// per-wave LDS: X stage (32*608*2) + H buffer (32*256*2) + 32 area floats
#define XS_BYTES   (32 * XROW * 2)        // 38912
#define H_BYTES    (32 * HID * 2)         // 16384
#define PER_WAVE_BYTES (XS_BYTES + H_BYTES + 128)   // 55424, 16B aligned

// ---------------------------------------------------------------------------
// Repack a KxN f32 weight matrix into per-lane WMMA B-fragment order (f16).
// Packed layout: tile (kt,nt) at ((kt*NT)+nt)*512 halves; within a tile,
// lane L owns 16 contiguous halves:  n = nt*16 + (L&15),
//                                    k = kt*32 + (L>=16 ? 16 : 0) + e
// (contiguous k order used here, same as A: lanes 0-15 hold K=0..15, lanes 16-31 hold K=16..31.)
// ---------------------------------------------------------------------------
__global__ void pack_weights_f16(const float* __restrict__ src, int K, int N,
                                 int KT, int NT, _Float16* __restrict__ dst) {
  int idx = blockIdx.x * blockDim.x + threadIdx.x;
  int total = KT * NT * 512;
  if (idx >= total) return;
  int e    = idx & 15;
  int lane = (idx >> 4) & 31;
  int tile = idx >> 9;
  int kt = tile / NT;
  int nt = tile - kt * NT;
  int n = nt * 16 + (lane & 15);
  int k = kt * 32 + ((lane >= 16) ? 16 : 0) + e;
  float v = (k < K && n < N) ? src[(size_t)k * N + n] : 0.f;
  dst[idx] = (_Float16)v;
}

__device__ __forceinline__ v16h join8(v8h lo, v8h hi) {
  return __builtin_shufflevector(lo, hi, 0,1,2,3,4,5,6,7,8,9,10,11,12,13,14,15);
}

// ---------------------------------------------------------------------------
// 2x2 block of 16x16 output tiles sharing A and B streams:
//   aMN = A[Mtile M] * B[Ntile N] + biasN
// A fragment per lane (contiguous k order, same as packed B):
//   lanes 0-15 : K = [0..15]   (two contiguous 16B LDS reads)
//   lanes 16-31: K = [16..31]
// ---------------------------------------------------------------------------
__device__ __forceinline__ void wmma_tile22(const _Float16* act, int stride,
                                            int ktiles,
                                            const _Float16* pwN0,
                                            const _Float16* pwN1,
                                            int tileStride, int lane,
                                            float bias0, float bias1,
                                            v8f& a00, v8f& a01,
                                            v8f& a10, v8f& a11) {
  a00 = (v8f){bias0, bias0, bias0, bias0, bias0, bias0, bias0, bias0};
  a10 = a00;
  a01 = (v8f){bias1, bias1, bias1, bias1, bias1, bias1, bias1, bias1};
  a11 = a01;
  const int m  = lane & 15;
  const int ka = (lane >= 16) ? 16 : 0;
  const _Float16* ar0 = act + (size_t)m * stride + ka;
  const _Float16* ar1 = act + (size_t)(m + 16) * stride + ka;
  const _Float16* bp0 = pwN0 + lane * 16;
  const _Float16* bp1 = pwN1 + lane * 16;
  for (int kt = 0; kt < ktiles; ++kt) {
    v16h b0 = join8(*(const v8h*)(bp0), *(const v8h*)(bp0 + 8));
    v16h b1 = join8(*(const v8h*)(bp1), *(const v8h*)(bp1 + 8));
    v16h A0 = join8(*(const v8h*)(ar0), *(const v8h*)(ar0 + 8));
    v16h A1 = join8(*(const v8h*)(ar1), *(const v8h*)(ar1 + 8));
    a00 = __builtin_amdgcn_wmma_f32_16x16x32_f16(false, A0, false, b0,
                                                 (short)0, a00, false, false);
    a01 = __builtin_amdgcn_wmma_f32_16x16x32_f16(false, A0, false, b1,
                                                 (short)0, a01, false, false);
    a10 = __builtin_amdgcn_wmma_f32_16x16x32_f16(false, A1, false, b0,
                                                 (short)0, a10, false, false);
    a11 = __builtin_amdgcn_wmma_f32_16x16x32_f16(false, A1, false, b1,
                                                 (short)0, a11, false, false);
    ar0 += 32;
    ar1 += 32;
    bp0 += tileStride;
    bp1 += tileStride;
  }
}

// One hidden layer for both M-tiles: actIn (32 x 32*ktiles) -> ReLU ->
// actOut (32 x 256, f16).  D layout: lane L, vgpr v -> (M = v + (L>=16?8:0),
// N = L&15).
__device__ __forceinline__ void mlp_hidden_layer(const _Float16* actIn,
                                                 int inStride, int ktiles,
                                                 const _Float16* pw,
                                                 const float* bias,
                                                 _Float16* actOut, int lane) {
  const int n0   = lane & 15;
  const int moff = (lane >= 16) ? 8 : 0;
  for (int nt = 0; nt < NT_H; nt += 2) {
    const float bi0 = bias[nt * 16 + n0];
    const float bi1 = bias[(nt + 1) * 16 + n0];
    v8f a00, a01, a10, a11;
    wmma_tile22(actIn, inStride, ktiles, pw + nt * 512, pw + (nt + 1) * 512,
                NT_H * 512, lane, bi0, bi1, a00, a01, a10, a11);
    _Float16* o0 = actOut + nt * 16 + n0;        // N-tile nt
    _Float16* o1 = o0 + 16;                      // N-tile nt+1
#pragma unroll
    for (int v = 0; v < 8; ++v) {
      const size_t r0 = (size_t)(v + moff) * HID;
      const size_t r1 = (size_t)(16 + v + moff) * HID;
      o0[r0] = (_Float16)fmaxf(a00[v], 0.f);
      o1[r0] = (_Float16)fmaxf(a01[v], 0.f);
      o0[r1] = (_Float16)fmaxf(a10[v], 0.f);
      o1[r1] = (_Float16)fmaxf(a11[v], 0.f);
    }
  }
}

// ---------------------------------------------------------------------------
// Main fused kernel. One wave = 32 rows = 8 queries x 4 shift variants, so
// the area-weighted blend is wave-local.  Row gr = qg*4 + s,
// s: 0=(-1,-1) 1=(-1,+1) 2=(+1,-1) 3=(+1,+1);  blend weight for pred s is
// area[3-s]/tot (reference swaps 0<->3 and 1<->2).
// ---------------------------------------------------------------------------
__global__ void liif_wmma_kernel(const float* __restrict__ feat,
                                 const float* __restrict__ coord,
                                 const float* __restrict__ cell,
                                 const _Float16* __restrict__ pw0,
                                 const _Float16* __restrict__ pw1,
                                 const _Float16* __restrict__ pw2,
                                 const _Float16* __restrict__ pw3,
                                 const _Float16* __restrict__ pw4,
                                 const float* __restrict__ bs0,
                                 const float* __restrict__ bs1,
                                 const float* __restrict__ bs2,
                                 const float* __restrict__ bs3,
                                 const float* __restrict__ bs4,
                                 float* __restrict__ out) {
  extern __shared__ char smem[];
  const int lane = threadIdx.x & 31;
  const int wave = threadIdx.x >> 5;
  const int tile = blockIdx.x * WAVES + wave;     // 32-row tile index
  char* wb = smem + (size_t)wave * PER_WAVE_BYTES;
  _Float16* Xs = (_Float16*)wb;                       // 32 x 608 f16
  _Float16* Ha = (_Float16*)(wb + XS_BYTES);          // 32 x 256 f16
  _Float16* Hb = Xs;                                  // aliases dead X stage
  float* areaS = (float*)(wb + XS_BYTES + H_BYTES);   // 32 floats

  // ---------------- stage 1: build 32 input rows + areas ----------------
  for (int mt = 0; mt < MT; ++mt) {
    const int m = mt * 16 + (lane & 15);   // local row 0..31
    const int h = lane >> 4;               // half (channel split)
    const int gr = tile * 32 + m;          // global row
    const int qg = gr >> 2;                // b*Q + q
    const int s  = gr & 3;
    const int b  = qg >> 13;               // Q = 8192
    const float* cq = coord + (size_t)qg * 18;
    const float sh0 = (((s & 2) ? 1.f : -1.f) / 48.f) + 1e-6f;
    const float sh1 = (((s & 1) ? 1.f : -1.f) / 48.f) + 1e-6f;
    float cen0 = fminf(fmaxf(cq[8] + sh0, -1.f + 1e-6f), 1.f - 1e-6f);
    float cen1 = fminf(fmaxf(cq[9] + sh1, -1.f + 1e-6f), 1.f - 1e-6f);
    int yi = (int)rintf((cen0 + 1.f) * 24.f - 0.5f);
    int xi = (int)rintf((cen1 + 1.f) * 24.f - 0.5f);
    yi = yi < 0 ? 0 : (yi > 47 ? 47 : yi);
    xi = xi < 0 ? 0 : (xi > 47 ? 47 : xi);
    const float qcr = (float)(2 * yi + 1) / 48.f - 1.f;
    const float qcc = (float)(2 * xi + 1) / 48.f - 1.f;
    const float* fb = feat + ((size_t)b * 64 + (size_t)h * 32) * 2304;
    _Float16* xr = Xs + (size_t)m * XROW;
    // 3x3 unfold gather: feature index = c*9 + (i*3+j)
    for (int p = 0; p < 9; ++p) {
      const int y = yi + p / 3 - 1;
      const int x = xi + p % 3 - 1;
      const bool ok = ((unsigned)y < 48u) && ((unsigned)x < 48u);
      const int off = y * 48 + x;
      for (int cc = 0; cc < 32; ++cc) {
        float v = ok ? fb[(size_t)cc * 2304 + off] : 0.f;
        xr[(h * 32 + cc) * 9 + p] = (_Float16)v;
      }
    }
    if (h == 0) {
      for (int p = 0; p < 9; ++p) {   // rel, scaled by [H*W, 1]
        xr[576 + 2 * p]     = (_Float16)((cq[2 * p]     - qcr) * 2304.f);
        xr[576 + 2 * p + 1] = (_Float16)((cq[2 * p + 1] - qcc));
      }
      const float* ce = cell + (size_t)qg * 2;
      xr[594] = (_Float16)(ce[0] * 48.f);
      xr[595] = (_Float16)(ce[1] * 48.f);
      for (int k = 596; k < 608; ++k) xr[k] = (_Float16)0.f;  // zero pad
      areaS[m] = fabsf((cq[0] - qcr) * 2304.f * (cq[1] - qcc)) + 1e-9f;
    }
  }
  __syncthreads();

  // ---------------- MLP: 4 hidden layers + head ----------------
  mlp_hidden_layer(Xs, XROW, KT0, pw0, bs0, Ha, lane);
  __syncthreads();
  mlp_hidden_layer(Ha, HID, KTH, pw1, bs1, Hb, lane);
  __syncthreads();
  mlp_hidden_layer(Hb, HID, KTH, pw2, bs2, Ha, lane);
  __syncthreads();
  mlp_hidden_layer(Ha, HID, KTH, pw3, bs3, Hb, lane);
  __syncthreads();

  // head (256 -> 27, padded to 32) + area blend + store
  {
    const int n0 = lane & 15;
    const float bi0 = bs4[n0];
    const float bi1 = (16 + n0 < 27) ? bs4[16 + n0] : 0.f;
    // loM = cols 0-15 of M-tile M, hiM = cols 16-31 of M-tile M
    v8f lo0, hi0, lo1, hi1;
    wmma_tile22(Hb, HID, KTH, pw4, pw4 + 512, 2 * 512, lane, bi0, bi1,
                lo0, hi0, lo1, hi1);
    const int moff = (lane >= 16) ? 8 : 0;
#pragma unroll
    for (int mt = 0; mt < MT; ++mt) {
      v8f accLo = (mt == 0) ? lo0 : lo1;
      v8f accHi = (mt == 0) ? hi0 : hi1;
      float wrow[8];
#pragma unroll
      for (int v = 0; v < 8; ++v) {
        const int m = mt * 16 + v + moff;      // local row 0..31
        const int qgl = m >> 2;
        const int s = m & 3;
        const float tot = areaS[qgl * 4] + areaS[qgl * 4 + 1] +
                          areaS[qgl * 4 + 2] + areaS[qgl * 4 + 3];
        wrow[v] = areaS[qgl * 4 + (3 - s)] / tot;
      }
      float sA0 = 0.f, sA1 = 0.f, sB0 = 0.f, sB1 = 0.f;
#pragma unroll
      for (int v = 0; v < 4; ++v) { sA0 += accLo[v] * wrow[v]; sA1 += accHi[v] * wrow[v]; }
#pragma unroll
      for (int v = 4; v < 8; ++v) { sB0 += accLo[v] * wrow[v]; sB1 += accHi[v] * wrow[v]; }
      const size_t qgA = (size_t)tile * 8 + mt * 4 + ((lane >= 16) ? 2 : 0);
      out[qgA * 27 + n0]       = sA0;
      out[(qgA + 1) * 27 + n0] = sB0;
      if (16 + n0 < 27) {
        out[qgA * 27 + 16 + n0]       = sA1;
        out[(qgA + 1) * 27 + 16 + n0] = sB1;
      }
    }
  }
}

// ---------------------------------------------------------------------------
extern "C" void kernel_launch(void* const* d_in, const int* in_sizes, int n_in,
                              void* d_out, int out_size, void* d_ws, size_t ws_size,
                              hipStream_t stream) {
  (void)in_sizes; (void)n_in; (void)out_size; (void)ws_size;
  const float* feat  = (const float*)d_in[0];
  const float* coord = (const float*)d_in[1];
  const float* cell  = (const float*)d_in[2];
  const float* w0 = (const float*)d_in[3];
  const float* b0 = (const float*)d_in[4];
  const float* w1 = (const float*)d_in[5];
  const float* b1 = (const float*)d_in[6];
  const float* w2 = (const float*)d_in[7];
  const float* b2 = (const float*)d_in[8];
  const float* w3 = (const float*)d_in[9];
  const float* b3 = (const float*)d_in[10];
  const float* w4 = (const float*)d_in[11];
  const float* b4 = (const float*)d_in[12];

  // workspace: packed f16 weights (deterministically rebuilt every call)
  _Float16* pw0 = (_Float16*)d_ws;          // 19*16*512 = 155648 halves
  _Float16* pw1 = pw0 + 19 * 16 * 512;      //  8*16*512 =  65536
  _Float16* pw2 = pw1 + 8 * 16 * 512;
  _Float16* pw3 = pw2 + 8 * 16 * 512;
  _Float16* pw4 = pw3 + 8 * 16 * 512;       //  8* 2*512 =   8192

  auto pack = [&](const float* src, int K, int N, int KT, int NT, _Float16* dst) {
    int total = KT * NT * 512;
    pack_weights_f16<<<(total + 255) / 256, 256, 0, stream>>>(src, K, N, KT, NT, dst);
  };
  pack(w0, 596, 256, 19, 16, pw0);
  pack(w1, 256, 256, 8, 16, pw1);
  pack(w2, 256, 256, 8, 16, pw2);
  pack(w3, 256, 256, 8, 16, pw3);
  pack(w4, 256, 27,  8, 2,  pw4);

  const size_t ldsBytes = (size_t)WAVES * PER_WAVE_BYTES;  // ~216.5 KB / WG
  (void)hipFuncSetAttribute((const void*)liif_wmma_kernel,
                            hipFuncAttributeMaxDynamicSharedMemorySize,
                            (int)ldsBytes);
  const int tiles  = (8 * 8192 * 4) / 32;   // 8192 32-row tiles
  const int blocks = tiles / WAVES;         // 2048
  liif_wmma_kernel<<<blocks, 32 * WAVES, ldsBytes, stream>>>(
      feat, coord, cell, pw0, pw1, pw2, pw3, pw4,
      b0, b1, b2, b3, b4, (float*)d_out);
}
